// Attention_3762391352071
// MI455X (gfx1250) — matrix operand order repaired and checked by reading, not compiled, not run on hardware
//
#include <hip/hip_runtime.h>
#include <hip/hip_bf16.h>

#define DIM    512
#define HEADS  8
#define DHEAD  64
#define SEQ    1024
#define BATCH  8
#define NTOK   (BATCH * SEQ)   /* 8192 */
#define INNER  512
#define SCALE_F 0.125f         /* 64^-0.5 */

typedef __attribute__((ext_vector_type(16))) __bf16 v16bf;
typedef __attribute__((ext_vector_type(8)))  __bf16 v8bf;
typedef __attribute__((ext_vector_type(8)))  float  v8f;
typedef __attribute__((ext_vector_type(4)))  float  f32x4;

static __device__ inline __bf16 f2bf(float f) {
  unsigned u = __builtin_bit_cast(unsigned, f);
  unsigned r = u + 0x7FFFu + ((u >> 16) & 1u);   // round-to-nearest-even
  unsigned short h = (unsigned short)(r >> 16);
  return __builtin_bit_cast(__bf16, h);
}

static __device__ inline v8f wmma_bf16(v16bf a, v16bf b, v8f c) {
  return __builtin_amdgcn_wmma_f32_16x16x32_bf16(
      /*neg_a=*/false, a, /*neg_b=*/false, b,
      /*c_mod=*/(short)0, c, /*reuse_a=*/false, /*reuse_b=*/false);
}

// ---- A fragment (16x32 MxK bf16) from bf16 row-major [.., ld] ----
// lane<16: K = {k0..k0+7, k0+16..k0+23}; lane>=16: +8 on both groups.
static __device__ inline v16bf load_a_bf16(const __bf16* A, int m0, int k0, int ld, int lane) {
  int row = m0 + (lane & 15);
  int kb  = k0 + ((lane >> 4) << 3);
  const __bf16* p = A + (size_t)row * ld + kb;
  v8bf lo = *(const v8bf*)(p);
  v8bf hi = *(const v8bf*)(p + 16);
  return __builtin_shufflevector(lo, hi, 0, 1, 2, 3, 4, 5, 6, 7,
                                 8, 9, 10, 11, 12, 13, 14, 15);
}

// ---- B fragment (32x16 KxN bf16) where B[k][n] = S[n0+n][k0+k], S bf16 row-major ----
// lane holds col n = n0 + lane%16, K = {k0..k0+7, k0+16..k0+23} (+8 for hi lanes).
static __device__ inline v16bf load_bT_bf16(const __bf16* S, int n0, int k0, int ld, int lane) {
  int n  = n0 + (lane & 15);
  int kb = k0 + ((lane >> 4) << 3);
  const __bf16* p = S + (size_t)n * ld + kb;
  v8bf lo = *(const v8bf*)(p);
  v8bf hi = *(const v8bf*)(p + 16);
  return __builtin_shufflevector(lo, hi, 0, 1, 2, 3, 4, 5, 6, 7,
                                 8, 9, 10, 11, 12, 13, 14, 15);
}

// Async memory -> LDS copy of one 16-byte chunk per lane (wave32: 512 B/instr).
// Tracked by ASYNCcnt; completion enforced with s_wait_asynccnt.
static __device__ inline void async_b128_to_lds(const void* gptr, const void* lptr) {
  unsigned lds = (unsigned)(size_t)lptr;                 // low 32 bits = LDS offset
  unsigned long long ga = (unsigned long long)(size_t)gptr;
  asm volatile("global_load_async_to_lds_b128 %0, %1, off"
               :: "v"(lds), "v"(ga) : "memory");
}

static __device__ inline void wait_async0() {
  asm volatile("s_wait_asynccnt 0" ::: "memory");
}

// ===================================================================
// Kernel 0: one-shot f32 -> bf16 conversion (memory-bound pre-pass).
// ===================================================================
__global__ __launch_bounds__(256) void cvt_bf16_kernel(
    const float* __restrict__ src, __bf16* __restrict__ dst, int n8) {
  int idx = blockIdx.x * blockDim.x + threadIdx.x;
  if (idx >= n8) return;
  const f32x4* s = (const f32x4*)src + (size_t)idx * 2;
  f32x4 a = s[0], b = s[1];
  v8bf o;
#pragma unroll
  for (int e = 0; e < 4; ++e) {
    o[e]     = f2bf(a[e]);
    o[4 + e] = f2bf(b[e]);
  }
  ((v8bf*)dst)[idx] = o;
}

// ===================================================================
// Kernel 1: fused QKV projection (all-bf16 hot loop).
// xbf[8192,512] @ {wq,wk,wv}^T -> q,k [b,h,n,64] bf16, vT [b,h,64,n] bf16.
// ===================================================================
__global__ __launch_bounds__(256) void qkv_kernel(
    const __bf16* __restrict__ xbf,
    const __bf16* __restrict__ wqb, const __bf16* __restrict__ wkb,
    const __bf16* __restrict__ wvb,
    __bf16* __restrict__ qws, __bf16* __restrict__ kws,
    __bf16* __restrict__ vtws) {
  int lane = threadIdx.x & 31;
  int wave = threadIdx.x >> 5;
  int gw = blockIdx.x * 8 + wave;          // 4096 waves
  int mt = gw >> 3;                        // 512 m-tiles
  int nc = gw & 7;                         // 8 n-chunks of 64
  int m0 = mt << 4;
  int n0 = nc << 6;

  v8f z = {0.f, 0.f, 0.f, 0.f, 0.f, 0.f, 0.f, 0.f};
  v8f acc[12];
#pragma unroll
  for (int i = 0; i < 12; ++i) acc[i] = z;

  const __bf16* W[3] = {wqb, wkb, wvb};
  for (int k0 = 0; k0 < DIM; k0 += 32) {
    v16bf a = load_a_bf16(xbf, m0, k0, DIM, lane);
#pragma unroll
    for (int w = 0; w < 3; ++w) {
#pragma unroll
      for (int nt = 0; nt < 4; ++nt) {
        v16bf b = load_bT_bf16(W[w], n0 + nt * 16, k0, DIM, lane);
        acc[w * 4 + nt] = wmma_bf16(a, b, acc[w * 4 + nt]);
      }
    }
  }

  int hioff = (lane >> 4) << 3;
#pragma unroll
  for (int nt = 0; nt < 4; ++nt) {
    int col = n0 + nt * 16 + (lane & 15);
    int h = col >> 6, d = col & 63;
#pragma unroll
    for (int r = 0; r < 8; ++r) {
      int row = m0 + r + hioff;
      int bb = row >> 10, nn = row & 1023;
      size_t qkidx = (((size_t)bb * HEADS + h) * SEQ + nn) * DHEAD + d;
      size_t vtidx = (((size_t)bb * HEADS + h) * DHEAD + d) * SEQ + nn;
      qws[qkidx]  = f2bf(acc[0 + nt][r]);
      kws[qkidx]  = f2bf(acc[4 + nt][r]);
      vtws[vtidx] = f2bf(acc[8 + nt][r]);
    }
  }
}

// ===================================================================
// Kernel 2: attention.  Block = 8 waves sharing one (b,h); wave w owns
// i-tile (ig*8+w).  K/V j-tiles staged cooperatively into LDS with
// double-buffered global_load_async_to_lds_b128; B fragments read from LDS.
// S = Q K^T, e = exp(clip(S*scale)), O = (e V) * 1/rowsum(e).
// ===================================================================
__global__ __launch_bounds__(256) void attn_kernel(
    const __bf16* __restrict__ qws, const __bf16* __restrict__ kws,
    const __bf16* __restrict__ vtws, __bf16* __restrict__ ows) {
  __shared__ __bf16 ldsK[2][32][64];       // j-tile of K  (rows j, cols d)  2x4 KB
  __shared__ __bf16 ldsV[2][64][32];       // j-tile of V^T (rows d, cols j) 2x4 KB
  __shared__ __bf16 els[8][16][32];        // per-wave 16x32 e-tile          8 KB

  int lane = threadIdx.x & 31;
  int wave = threadIdx.x >> 5;
  int bh = blockIdx.x >> 3;                // 64 (b,h) pairs
  int ig = blockIdx.x & 7;                 // 8 i-tile groups
  int i0 = ((ig << 3) + wave) << 4;
  int hioff = (lane >> 4) << 3;

  const __bf16* qbh  = qws + (size_t)bh * SEQ * DHEAD;
  const __bf16* kbh  = kws + (size_t)bh * SEQ * DHEAD;
  const __bf16* vtbh = vtws + (size_t)bh * DHEAD * SEQ;

  // per-wave 512B slices of the 4KB K and V tiles
  int kr = (wave << 2) + (lane >> 3);      // K row (j-rel), 0..31
  int kc = (lane & 7) << 3;                // K col (d), 16B chunks
  int vr = (wave << 3) + (lane >> 2);      // V row (d), 0..63
  int vc = (lane & 3) << 3;                // V col (j-rel), 16B chunks

  // stage j-tile starting at j0 into buffer buf
  auto stage = [&](int j0, int buf) {
    async_b128_to_lds(kbh + (size_t)(j0 + kr) * DHEAD + kc, &ldsK[buf][kr][kc]);
    async_b128_to_lds(vtbh + (size_t)vr * SEQ + (j0 + vc), &ldsV[buf][vr][vc]);
  };

  v16bf aq0 = load_a_bf16(qbh, i0, 0, DHEAD, lane);
  v16bf aq1 = load_a_bf16(qbh, i0, 32, DHEAD, lane);

  v8f z = {0.f, 0.f, 0.f, 0.f, 0.f, 0.f, 0.f, 0.f};
  v8f acc[4];
#pragma unroll
  for (int i = 0; i < 4; ++i) acc[i] = z;
  float psum[8] = {0.f, 0.f, 0.f, 0.f, 0.f, 0.f, 0.f, 0.f};

  stage(0, 0);
  wait_async0();
  __syncthreads();

  for (int j0 = 0; j0 < SEQ; j0 += 32) {
    int cur = (j0 >> 5) & 1;
    if (j0 + 32 < SEQ) stage(j0 + 32, cur ^ 1);   // prefetch next tile (other buffer)

    const __bf16* Kt = &ldsK[cur][0][0];          // [32][64]
    const __bf16* Vt = &ldsV[cur][0][0];          // [64][32]

#pragma unroll
    for (int t = 0; t < 2; ++t) {
      // B[k=d][n=j] = Kt[jrel][d] : transposed-source fragment, ld=64
      v16bf bk0 = load_bT_bf16(Kt, t * 16, 0, 64, lane);
      v16bf bk1 = load_bT_bf16(Kt, t * 16, 32, 64, lane);
      v8f s = z;
      s = wmma_bf16(aq0, bk0, s);
      s = wmma_bf16(aq1, bk1, s);
#pragma unroll
      for (int r = 0; r < 8; ++r) {
        float e = __expf(fminf(fmaxf(s[r] * SCALE_F, 1e-6f), 1.0f));
        psum[r] += e;
        els[wave][r + hioff][t * 16 + (lane & 15)] = f2bf(e);
      }
    }
    // intra-wave exchange through LDS: DS ops are in-order per wave, no barrier
    const __bf16* ep = &els[wave][lane & 15][hioff];
    v8bf lo = *(const v8bf*)(ep);
    v8bf hi = *(const v8bf*)(ep + 16);
    v16bf aat = __builtin_shufflevector(lo, hi, 0, 1, 2, 3, 4, 5, 6, 7,
                                        8, 9, 10, 11, 12, 13, 14, 15);
#pragma unroll
    for (int dt = 0; dt < 4; ++dt) {
      // B[k=jrel][n=d] = Vt[d][jrel] : transposed-source fragment, ld=32
      v16bf bv = load_bT_bf16(Vt, dt * 16, 0, 32, lane);
      acc[dt] = wmma_bf16(aat, bv, acc[dt]);
    }

    wait_async0();        // prefetched tile resident
    __syncthreads();      // all waves done with cur buffer; next buffer visible
  }

  // reduce row sums across the 16-lane half (rows match O-fragment half)
#pragma unroll
  for (int r = 0; r < 8; ++r) {
#pragma unroll
    for (int m = 1; m < 16; m <<= 1) psum[r] += __shfl_xor(psum[r], m, 32);
  }

  int bb = bh >> 3, hh = bh & 7;
#pragma unroll
  for (int dt = 0; dt < 4; ++dt) {
    int col = hh * 64 + dt * 16 + (lane & 15);
#pragma unroll
    for (int r = 0; r < 8; ++r) {
      int row = i0 + r + hioff;
      ows[((size_t)bb * SEQ + row) * INNER + col] = f2bf(acc[dt][r] / psum[r]);
    }
  }
}

// ===================================================================
// Kernel 3: output projection.  ows[8192,512] bf16 @ wo^T + bo -> f32 out.
// ===================================================================
__global__ __launch_bounds__(256) void outproj_kernel(
    const __bf16* __restrict__ ows, const __bf16* __restrict__ wob,
    const float* __restrict__ bo, float* __restrict__ out) {
  int lane = threadIdx.x & 31;
  int wave = threadIdx.x >> 5;
  int gw = blockIdx.x * 8 + wave;
  int mt = gw >> 3, nc = gw & 7;
  int m0 = mt << 4, n0 = nc << 6;

  v8f z = {0.f, 0.f, 0.f, 0.f, 0.f, 0.f, 0.f, 0.f};
  v8f acc[4];
#pragma unroll
  for (int i = 0; i < 4; ++i) acc[i] = z;

  for (int k0 = 0; k0 < INNER; k0 += 32) {
    v16bf a = load_a_bf16(ows, m0, k0, INNER, lane);
#pragma unroll
    for (int nt = 0; nt < 4; ++nt) {
      v16bf b = load_bT_bf16(wob, n0 + nt * 16, k0, INNER, lane);
      acc[nt] = wmma_bf16(a, b, acc[nt]);
    }
  }

  int hioff = (lane >> 4) << 3;
#pragma unroll
  for (int nt = 0; nt < 4; ++nt) {
    int col = n0 + nt * 16 + (lane & 15);
    float bias = bo[col];
#pragma unroll
    for (int r = 0; r < 8; ++r) {
      int row = m0 + r + hioff;
      out[(size_t)row * DIM + col] = acc[nt][r] + bias;
    }
  }
}

extern "C" void kernel_launch(void* const* d_in, const int* in_sizes, int n_in,
                              void* d_out, int out_size, void* d_ws, size_t ws_size,
                              hipStream_t stream) {
  const float* x  = (const float*)d_in[0];
  const float* wq = (const float*)d_in[1];
  const float* wk = (const float*)d_in[2];
  const float* wv = (const float*)d_in[3];
  const float* wo = (const float*)d_in[4];
  const float* bo = (const float*)d_in[5];
  float* out = (float*)d_out;

  const size_t BUF  = (size_t)NTOK * INNER;   // 8192*512 elements
  const size_t WBUF = (size_t)DIM * DIM;      // 512*512 elements
  __bf16* qws  = (__bf16*)d_ws;
  __bf16* kws  = qws + BUF;
  __bf16* vtws = kws + BUF;
  __bf16* ows  = vtws + BUF;
  __bf16* xbf  = ows + BUF;
  __bf16* wqb  = xbf + BUF;
  __bf16* wkb  = wqb + WBUF;
  __bf16* wvb  = wkb + WBUF;
  __bf16* wob  = wvb + WBUF;

  // pre-pass: f32 -> bf16 once (memory-bound, ~30 MB of traffic)
  {
    int n8 = (int)(BUF / 8);
    cvt_bf16_kernel<<<(n8 + 255) / 256, 256, 0, stream>>>(x, xbf, n8);
    int w8 = (int)(WBUF / 8);
    cvt_bf16_kernel<<<(w8 + 255) / 256, 256, 0, stream>>>(wq, wqb, w8);
    cvt_bf16_kernel<<<(w8 + 255) / 256, 256, 0, stream>>>(wk, wkb, w8);
    cvt_bf16_kernel<<<(w8 + 255) / 256, 256, 0, stream>>>(wv, wvb, w8);
    cvt_bf16_kernel<<<(w8 + 255) / 256, 256, 0, stream>>>(wo, wob, w8);
  }

  qkv_kernel<<<512, 256, 0, stream>>>(xbf, wqb, wkb, wvb, qws, kws, vtws);
  attn_kernel<<<512, 256, 0, stream>>>(qws, kws, vtws, ows);
  outproj_kernel<<<512, 256, 0, stream>>>(ows, wob, bo, out);
}
